// EVAttnProcessor2_0_W_IP_55164559950353
// MI455X (gfx1250) — compile-verified, matrix operand order checked by reading, not run on hardware
//
#include <hip/hip_runtime.h>

typedef __bf16 bf16_t;
typedef __attribute__((ext_vector_type(8)))  bf16_t v8bf;
typedef __attribute__((ext_vector_type(16))) bf16_t v16bf;
typedef __attribute__((ext_vector_type(8)))  float  v8f;

union BFrag { v16bf v; v8bf h[2]; };

static __device__ __forceinline__ v8f wmma_bf16(const BFrag& a, const BFrag& b, v8f c) {
  return __builtin_amdgcn_wmma_f32_16x16x32_bf16(false, a.v, false, b.v, (short)0, c,
                                                 false, false);
}
static __device__ __forceinline__ v8f zero_v8f() {
  v8f z = {0.f, 0.f, 0.f, 0.f, 0.f, 0.f, 0.f, 0.f};
  return z;
}
static __device__ __forceinline__ v8bf zero_v8bf() {
  bf16_t z = (bf16_t)0.0f;
  v8bf v = {z, z, z, z, z, z, z, z};
  return v;
}

// ---------------- conversion / staging kernels ----------------

__global__ void cvt_bf16_kernel(const float* __restrict__ in, bf16_t* __restrict__ out, long n) {
  long i = (long)blockIdx.x * blockDim.x + threadIdx.x;
  long stride = (long)gridDim.x * blockDim.x;
  for (; i < n; i += stride) out[i] = (bf16_t)in[i];
}

// out[c*R + r] = (bf16) in[r*C + c]   (weight [K x N] -> [N x K] bf16)
__global__ void transpose_bf16_kernel(const float* __restrict__ in, bf16_t* __restrict__ out,
                                      int R, int C) {
  long n = (long)R * C;
  long i = (long)blockIdx.x * blockDim.x + threadIdx.x;
  long stride = (long)gridDim.x * blockDim.x;
  for (; i < n; i += stride) {
    int r = (int)(i / C);
    int c = (int)(i % C);
    out[(long)c * R + r] = (bf16_t)in[i];
  }
}

// encoder_hidden_states [2][101][2048] f32 -> zero padded segment [Mpad][2048] bf16,
// rows laid out as b*Lpad + l, rows with l>=L or >= 2*Lpad are zero.
__global__ void gather_seg_kernel(const float* __restrict__ ehs, bf16_t* __restrict__ out,
                                  int seg_off, int L, int Lpad, int Mpad) {
  long n = (long)Mpad * 2048;
  long i = (long)blockIdx.x * blockDim.x + threadIdx.x;
  long stride = (long)gridDim.x * blockDim.x;
  for (; i < n; i += stride) {
    int c = (int)(i % 2048);
    int r = (int)(i / 2048);
    int b = r / Lpad;
    int l = r % Lpad;
    bool valid = (b < 2) && (l < L);
    out[i] = valid ? (bf16_t)ehs[((long)b * 101 + seg_off + l) * 2048 + c] : (bf16_t)0.0f;
  }
}

// ---------------- tiled bf16 WMMA GEMM (fully tile-aligned: M,N % 128 == 0, K % 32 == 0)
// C[M x N] = A[M x K] * B[K x N], Bt given as [N x K] (K-contiguous).
// MODE 0: store bf16 row-major [M x N]
// MODE 1: store bf16 transposed [N x M]
// MODE 2: store f32 row-major [M x N] = acc + bias[n] + resid[m*N+n]
template <int MODE>
__global__ __launch_bounds__(256, 1)
void gemm_bf16_kernel(const bf16_t* __restrict__ A, const bf16_t* __restrict__ Bt,
                      void* __restrict__ Cout, int M, int N, int K,
                      const float* __restrict__ bias, const float* __restrict__ resid) {
  __shared__ __align__(16) bf16_t lA[128 * 32];
  __shared__ __align__(16) bf16_t lB[128 * 32];

  const int t = threadIdx.x;
  const int lane = t & 31;
  const int wave = t >> 5;
  const int wm = wave & 3;   // 0..3 -> 32-row slab
  const int wn = wave >> 2;  // 0..1 -> 64-col slab
  const int ln15 = lane & 15;
  const int lh = lane >> 4;
  const int m0 = blockIdx.y * 128;
  const int n0 = blockIdx.x * 128;

  // global/LDS staging addresses: each thread moves 2 rows x 8 bf16 of A and of Bt
  const int lrow = t >> 2;       // 0..63
  const int lch  = (t & 3) * 8;  // chunk within 32-wide K slice
  const bf16_t* gA0 = A  + (long)(m0 + lrow)      * K + lch;
  const bf16_t* gA1 = A  + (long)(m0 + lrow + 64) * K + lch;
  const bf16_t* gB0 = Bt + (long)(n0 + lrow)      * K + lch;
  const bf16_t* gB1 = Bt + (long)(n0 + lrow + 64) * K + lch;
  bf16_t* sA0 = lA + lrow * 32 + lch;
  bf16_t* sA1 = lA + (lrow + 64) * 32 + lch;
  bf16_t* sB0 = lB + lrow * 32 + lch;
  bf16_t* sB1 = lB + (lrow + 64) * 32 + lch;

  v8f acc[2][4];
#pragma unroll
  for (int im = 0; im < 2; im++)
#pragma unroll
    for (int in_ = 0; in_ < 4; in_++) acc[im][in_] = zero_v8f();

  // software pipeline: preload K-slice 0
  v8bf ra0 = *(const v8bf*)gA0, ra1 = *(const v8bf*)gA1;
  v8bf rb0 = *(const v8bf*)gB0, rb1 = *(const v8bf*)gB1;

  for (int k0 = 0; k0 < K; k0 += 32) {
    __syncthreads();            // previous compute done, LDS reusable
    *(v8bf*)sA0 = ra0;
    *(v8bf*)sA1 = ra1;
    *(v8bf*)sB0 = rb0;
    *(v8bf*)sB1 = rb1;
    __syncthreads();            // tile visible

    // issue next K-slice's global loads (clamped, branch-free) to overlap with WMMAs
    int kn = (k0 + 32 < K) ? (k0 + 32) : 0;
    ra0 = *(const v8bf*)(gA0 + kn);
    ra1 = *(const v8bf*)(gA1 + kn);
    rb0 = *(const v8bf*)(gB0 + kn);
    rb1 = *(const v8bf*)(gB1 + kn);

    const int kb = lh * 8;
    BFrag af[2], bfm[4];
#pragma unroll
    for (int im = 0; im < 2; im++) {
      int r = wm * 32 + im * 16 + ln15;
      af[im].h[0] = *(const v8bf*)(lA + r * 32 + kb);
      af[im].h[1] = *(const v8bf*)(lA + r * 32 + kb + 16);
    }
#pragma unroll
    for (int in_ = 0; in_ < 4; in_++) {
      int c = wn * 64 + in_ * 16 + ln15;
      bfm[in_].h[0] = *(const v8bf*)(lB + c * 32 + kb);
      bfm[in_].h[1] = *(const v8bf*)(lB + c * 32 + kb + 16);
    }
#pragma unroll
    for (int im = 0; im < 2; im++)
#pragma unroll
      for (int in_ = 0; in_ < 4; in_++)
        acc[im][in_] = wmma_bf16(af[im], bfm[in_], acc[im][in_]);
  }

  // epilogue: lane holds rows j+8*lh, col ln15 of each 16x16 tile (no bounds needed)
#pragma unroll
  for (int im = 0; im < 2; im++) {
#pragma unroll
    for (int in_ = 0; in_ < 4; in_++) {
      const int n = n0 + wn * 64 + in_ * 16 + ln15;
      float bn = 0.f;
      if (MODE == 2) bn = bias[n];
#pragma unroll
      for (int j = 0; j < 8; j++) {
        const int m = m0 + wm * 32 + im * 16 + j + 8 * lh;
        const float v = acc[im][in_][j];
        if (MODE == 2) {
          ((float*)Cout)[(long)m * N + n] = v + bn + resid[(long)m * N + n];
        } else if (MODE == 1) {
          ((bf16_t*)Cout)[(long)n * M + m] = (bf16_t)v;
        } else {
          ((bf16_t*)Cout)[(long)m * N + n] = (bf16_t)v;
        }
      }
    }
  }
}

// ---------------- fused 4-branch attention ----------------
// One wave handles a 16-query tile of one (b, h). Q,K bf16 row-major [rows][1280],
// V given transposed [1280][Mp] so its WMMA B-fragments are key-contiguous.
__global__ __launch_bounds__(256, 1)
void attn_kernel(const bf16_t* __restrict__ Q,
                 const bf16_t* __restrict__ K0, const bf16_t* __restrict__ K1,
                 const bf16_t* __restrict__ K2, const bf16_t* __restrict__ K3,
                 const bf16_t* __restrict__ V0, const bf16_t* __restrict__ V1,
                 const bf16_t* __restrict__ V2, const bf16_t* __restrict__ V3,
                 bf16_t* __restrict__ Out) {
  __shared__ __align__(16) bf16_t pbuf[8][16 * 96];

  const int t = threadIdx.x;
  const int lane = t & 31;
  const int wave = t >> 5;
  const int ln15 = lane & 15;
  const int lh = lane >> 4;

  int tile = blockIdx.x * 8 + wave;   // 0..10239
  int qt = tile & 255;                // 256 q-tiles of 16
  int rest = tile >> 8;               // 0..39
  int h = rest % 20;
  int b = rest / 20;

  // Q A-fragments (16 x 64 -> two 16x32 frags)
  long qrow = ((long)b * 4096 + qt * 16 + ln15) * 1280 + h * 64;
  BFrag qf[2];
#pragma unroll
  for (int dk = 0; dk < 2; dk++) {
    qf[dk].h[0] = *(const v8bf*)(Q + qrow + dk * 32 + lh * 8);
    qf[dk].h[1] = *(const v8bf*)(Q + qrow + dk * 32 + lh * 8 + 16);
  }

  const bf16_t* Ks[4] = {K0, K1, K2, K3};
  const bf16_t* Vs[4] = {V0, V1, V2, V3};
  const int Lr[4] = {77, 4, 4, 16};
  const int Lp[4] = {96, 32, 32, 32};
  const int Mp[4] = {256, 128, 128, 128};  // padded leading dim of V (keys)
  const int NT[4] = {5, 1, 1, 1};

  v8f oacc[4];
#pragma unroll
  for (int dt = 0; dt < 4; dt++) oacc[dt] = zero_v8f();

  bf16_t* pb = &pbuf[wave][0];

#pragma unroll
  for (int s = 0; s < 4; s++) {
    const int L = Lr[s], Lpad = Lp[s], nt = NT[s];
    const int Mtot = Mp[s];
    const bf16_t* Kp = Ks[s];
    const bf16_t* Vp = Vs[s];

    v8f sc[5];
    for (int tt = 0; tt < nt; tt++) {
      long krow = ((long)b * Lpad + tt * 16 + ln15) * 1280 + h * 64;
      BFrag kf0, kf1;
      kf0.h[0] = *(const v8bf*)(Kp + krow + lh * 8);
      kf0.h[1] = *(const v8bf*)(Kp + krow + lh * 8 + 16);
      kf1.h[0] = *(const v8bf*)(Kp + krow + 32 + lh * 8);
      kf1.h[1] = *(const v8bf*)(Kp + krow + 32 + lh * 8 + 16);
      v8f s0 = wmma_bf16(qf[0], kf0, zero_v8f());
      s0 = wmma_bf16(qf[1], kf1, s0);
      int key = tt * 16 + ln15;
#pragma unroll
      for (int j = 0; j < 8; j++)
        s0[j] = (key < L) ? s0[j] * 0.125f : -1e30f;
      sc[tt] = s0;
    }

    // per-segment softmax over its keys; each row's 16 cols live across a 16-lane half
    float mx[8], sm[8];
#pragma unroll
    for (int j = 0; j < 8; j++) {
      float m = sc[0][j];
      for (int tt = 1; tt < nt; tt++) m = fmaxf(m, sc[tt][j]);
      mx[j] = m;
    }
#pragma unroll
    for (int j = 0; j < 8; j++)
      for (int msk = 1; msk < 16; msk <<= 1)
        mx[j] = fmaxf(mx[j], __shfl_xor(mx[j], msk, 32));
#pragma unroll
    for (int j = 0; j < 8; j++) sm[j] = 0.f;
    for (int tt = 0; tt < nt; tt++) {
#pragma unroll
      for (int j = 0; j < 8; j++) {
        float e = __expf(sc[tt][j] - mx[j]);
        sc[tt][j] = e;
        sm[j] += e;
      }
    }
#pragma unroll
    for (int j = 0; j < 8; j++)
      for (int msk = 1; msk < 16; msk <<= 1)
        sm[j] += __shfl_xor(sm[j], msk, 32);
#pragma unroll
    for (int j = 0; j < 8; j++) sm[j] = 1.f / sm[j];

    // stage P to LDS (zero-fill padded columns), converting C-layout -> row-major
    v8bf zv = zero_v8bf();
#pragma unroll
    for (int i = 0; i < 6; i++)
      *(v8bf*)(pb + lane * 48 + i * 8) = zv;
    for (int tt = 0; tt < nt; tt++) {
#pragma unroll
      for (int j = 0; j < 8; j++)
        pb[(j + 8 * lh) * 96 + tt * 16 + ln15] = (bf16_t)(sc[tt][j] * sm[j]);
    }

    // P (16 x Lpad) @ V (Lpad x 64)
    const int npf = Lpad / 32;
    for (int pf = 0; pf < npf; pf++) {
      BFrag pfr;
      int pr = ln15 * 96 + pf * 32 + lh * 8;
      pfr.h[0] = *(const v8bf*)(pb + pr);
      pfr.h[1] = *(const v8bf*)(pb + pr + 16);
#pragma unroll
      for (int dt = 0; dt < 4; dt++) {
        BFrag vf;
        long vrow = ((long)h * 64 + dt * 16 + ln15) * Mtot + b * Lpad + pf * 32 + lh * 8;
        vf.h[0] = *(const v8bf*)(Vp + vrow);
        vf.h[1] = *(const v8bf*)(Vp + vrow + 16);
        oacc[dt] = wmma_bf16(pfr, vf, oacc[dt]);
      }
    }
  }

  // store merged-head attention output as bf16 for the out-proj GEMM
#pragma unroll
  for (int dt = 0; dt < 4; dt++) {
#pragma unroll
    for (int j = 0; j < 8; j++) {
      long orow = ((long)b * 4096 + qt * 16 + j + 8 * lh) * 1280 + h * 64 + dt * 16 + ln15;
      Out[orow] = (bf16_t)oacc[dt][j];
    }
  }
}

// ---------------- host-side orchestration ----------------

extern "C" void kernel_launch(void* const* d_in, const int* in_sizes, int n_in,
                              void* d_out, int out_size, void* d_ws, size_t ws_size,
                              hipStream_t stream) {
  (void)in_sizes; (void)n_in; (void)out_size; (void)ws_size;

  const float* hs   = (const float*)d_in[0];   // [2][4096][1280]
  const float* ehs  = (const float*)d_in[1];   // [2][101][2048]
  const float* wq   = (const float*)d_in[2];   // [1280][1280]
  const float* wk[4] = {(const float*)d_in[3], (const float*)d_in[5],
                        (const float*)d_in[7], (const float*)d_in[9]};
  const float* wv[4] = {(const float*)d_in[4], (const float*)d_in[6],
                        (const float*)d_in[8], (const float*)d_in[10]};
  const float* wout = (const float*)d_in[11];  // [1280][1280]
  const float* bias = (const float*)d_in[12];  // [1280]

  const int Lr[4] = {77, 4, 4, 16};
  const int Lp[4] = {96, 32, 32, 32};
  const int Mp[4] = {256, 128, 128, 128};  // 128-aligned padded row counts
  const int Soff[4] = {0, 77, 81, 85};
  const long MR = 8192;   // B*S
  const int  C  = 1280;
  const int  CA = 2048;

  char* ws = (char*)d_ws;
  size_t off = 0;
  auto alloc = [&](size_t bytes) -> void* {
    void* p = ws + off;
    off = (off + bytes + 255) & ~(size_t)255;
    return p;
  };

  bf16_t* hs_bf   = (bf16_t*)alloc(MR * C * 2);
  bf16_t* q_bf    = (bf16_t*)alloc(MR * C * 2);
  bf16_t* attn_bf = (bf16_t*)alloc(MR * C * 2);
  bf16_t* wq_t    = (bf16_t*)alloc((size_t)C * C * 2);
  bf16_t* wout_t  = (bf16_t*)alloc((size_t)C * C * 2);
  bf16_t *wk_t[4], *wv_t[4], *ehs_seg[4], *kseg[4], *vtseg[4];
  for (int s = 0; s < 4; s++) {
    wk_t[s]    = (bf16_t*)alloc((size_t)C * CA * 2);
    wv_t[s]    = (bf16_t*)alloc((size_t)C * CA * 2);
    ehs_seg[s] = (bf16_t*)alloc((size_t)Mp[s] * CA * 2);
    kseg[s]    = (bf16_t*)alloc((size_t)Mp[s] * C * 2);
    vtseg[s]   = (bf16_t*)alloc((size_t)C * Mp[s] * 2);
  }

  // stage 1: precision staging
  cvt_bf16_kernel<<<4096, 256, 0, stream>>>(hs, hs_bf, MR * C);
  transpose_bf16_kernel<<<2048, 256, 0, stream>>>(wq, wq_t, C, C);
  transpose_bf16_kernel<<<2048, 256, 0, stream>>>(wout, wout_t, C, C);
  for (int s = 0; s < 4; s++) {
    transpose_bf16_kernel<<<2048, 256, 0, stream>>>(wk[s], wk_t[s], CA, C);
    transpose_bf16_kernel<<<2048, 256, 0, stream>>>(wv[s], wv_t[s], CA, C);
    gather_seg_kernel<<<512, 256, 0, stream>>>(ehs, ehs_seg[s], Soff[s], Lr[s], Lp[s], Mp[s]);
  }

  // stage 2: projections (all bf16 WMMA, fully tile-aligned)
  gemm_bf16_kernel<0><<<dim3(10, 64), 256, 0, stream>>>(
      hs_bf, wq_t, q_bf, (int)MR, C, C, nullptr, nullptr);
  for (int s = 0; s < 4; s++) {
    dim3 g(10, Mp[s] / 128);
    gemm_bf16_kernel<0><<<g, 256, 0, stream>>>(
        ehs_seg[s], wk_t[s], kseg[s], Mp[s], C, CA, nullptr, nullptr);
    gemm_bf16_kernel<1><<<g, 256, 0, stream>>>(
        ehs_seg[s], wv_t[s], vtseg[s], Mp[s], C, CA, nullptr, nullptr);  // transposed V
  }

  // stage 3: fused 4-branch attention (10240 wave-tiles, 8 waves/block)
  attn_kernel<<<1280, 256, 0, stream>>>(q_bf,
                                        kseg[0], kseg[1], kseg[2], kseg[3],
                                        vtseg[0], vtseg[1], vtseg[2], vtseg[3],
                                        attn_bf);

  // stage 4: out projection + bias + residual -> f32 output
  gemm_bf16_kernel<2><<<dim3(10, 64), 256, 0, stream>>>(
      attn_bf, wout_t, d_out, (int)MR, C, C, bias, hs);
}
